// TixyNet_23012434772548
// MI455X (gfx1250) — matrix operand order repaired and checked by reading, not compiled, not run on hardware
//
#include <hip/hip_runtime.h>
#include <hip/hip_bf16.h>
#include <math.h>

typedef __attribute__((ext_vector_type(16))) _Float16 v16h;
typedef __attribute__((ext_vector_type(8)))  _Float16 v8h;
typedef __attribute__((ext_vector_type(8)))  float    v8f;

#define BN_EPS 1e-5f
#define BATCH 2048
#define CCH   256
#define PADW  10
#define PADHW 100

union AFrag { v16h v; v8h h[2]; };

// ---------------------------------------------------------------------------
// Generic explicit GEMM:  out[M,N] = relu/id( A[M,K] * Bw[N,K]^T + bias[N] )
// A row-major f16 (K contiguous, K%32==0), Bw stored [N][K] f16.
// MODE 0: f16 relu -> out[row*N+col]
// MODE 1: f16 relu -> padded NHWC interior (rows are b*64+y*8+x, OH=OW=8)
// MODE 2: f32 (no relu) -> out[row*N+col]
// ---------------------------------------------------------------------------
template<int MODE>
__global__ __launch_bounds__(32)
void gemm_f16(const _Float16* __restrict__ A, const _Float16* __restrict__ Bw,
              const float* __restrict__ bias, void* __restrict__ out,
              int N, int K)
{
    const int lane  = threadIdx.x;
    const int mtile = blockIdx.x * 32;
    const int ntile = blockIdx.y * 64;
    const int lr = lane & 15;
    const int hi = lane >> 4;

    v8f acc[2][4];
    #pragma unroll
    for (int i = 0; i < 2; i++)
        #pragma unroll
        for (int j = 0; j < 4; j++) {
            v8f z = {0.f,0.f,0.f,0.f,0.f,0.f,0.f,0.f};
            acc[i][j] = z;
        }

    for (int kk = 0; kk < K; kk += 32) {
        AFrag a[2];
        #pragma unroll
        for (int mt = 0; mt < 2; mt++) {
            const _Float16* p = A + (size_t)(mtile + mt * 16 + lr) * K + kk + hi * 16;
            a[mt].h[0] = *(const v8h*)p;
            a[mt].h[1] = *(const v8h*)(p + 8);
        }
        #pragma unroll
        for (int nt = 0; nt < 4; nt++) {
            const _Float16* q = Bw + (size_t)(ntile + nt * 16 + lr) * K + kk + hi * 16;
            v16h bv = *(const v16h*)q;
            #pragma unroll
            for (int mt = 0; mt < 2; mt++)
                acc[mt][nt] = __builtin_amdgcn_wmma_f32_16x16x32_f16(
                    false, a[mt].v, false, bv, (short)0, acc[mt][nt], false, false);
        }
    }

    #pragma unroll
    for (int mt = 0; mt < 2; mt++)
        #pragma unroll
        for (int nt = 0; nt < 4; nt++)
            #pragma unroll
            for (int r = 0; r < 8; r++) {
                int row = mtile + mt * 16 + r + hi * 8;
                int col = ntile + nt * 16 + lr;
                float vv = acc[mt][nt][r] + bias[col];
                if (MODE == 2) {
                    ((float*)out)[(size_t)row * N + col] = vv;
                } else {
                    vv = fmaxf(vv, 0.0f);
                    if (MODE == 0) {
                        ((_Float16*)out)[(size_t)row * N + col] = (_Float16)vv;
                    } else {
                        int b = row >> 6, rem = row & 63, y = rem >> 3, x = rem & 7;
                        ((_Float16*)out)[((size_t)(b * PADHW + (y + 1) * PADW + (x + 1))) * CCH + col]
                            = (_Float16)vv;
                    }
                }
            }
}

// ---------------------------------------------------------------------------
// Implicit-GEMM 3x3 conv over padded NHWC f16 input [B][10][10][256].
// Weights Bw stored [OC=256][K=2304], k = (dy*3+dx)*256 + ic.
// MODE 0: SAME  (OH=OW=8, store padded interior of out)
// MODE 1: VALID (OH=OW=6, store compact [B][36][256])
// ---------------------------------------------------------------------------
template<int MODE>
__global__ __launch_bounds__(32)
void conv_gemm(const _Float16* __restrict__ In, const _Float16* __restrict__ Bw,
               const float* __restrict__ bias, _Float16* __restrict__ out)
{
    const int OH  = (MODE == 0) ? 8 : 6;
    const int OW  = OH;
    const int OFF = (MODE == 0) ? 0 : 1;
    const int K   = 9 * CCH;

    const int lane  = threadIdx.x;
    const int mtile = blockIdx.x * 32;
    const int ntile = blockIdx.y * 64;
    const int lr = lane & 15;
    const int hi = lane >> 4;

    int rb[2], ry[2], rx[2];
    #pragma unroll
    for (int mt = 0; mt < 2; mt++) {
        int row = mtile + mt * 16 + lr;
        rb[mt] = row / (OH * OW);
        int rem = row % (OH * OW);
        ry[mt] = rem / OW;
        rx[mt] = rem % OW;
    }

    v8f acc[2][4];
    #pragma unroll
    for (int i = 0; i < 2; i++)
        #pragma unroll
        for (int j = 0; j < 4; j++) {
            v8f z = {0.f,0.f,0.f,0.f,0.f,0.f,0.f,0.f};
            acc[i][j] = z;
        }

    for (int kc = 0; kc < 72; kc++) {
        int tap = kc >> 3;
        int icb = (kc & 7) * 32;
        int dy = tap / 3, dx = tap - dy * 3;

        AFrag a[2];
        #pragma unroll
        for (int mt = 0; mt < 2; mt++) {
            int py = ry[mt] + dy + OFF;
            int px = rx[mt] + dx + OFF;
            const _Float16* p = In + ((size_t)(rb[mt] * PADHW + py * PADW + px)) * CCH
                                + icb + hi * 16;
            a[mt].h[0] = *(const v8h*)p;
            a[mt].h[1] = *(const v8h*)(p + 8);
        }
        int kk = tap * CCH + icb;
        #pragma unroll
        for (int nt = 0; nt < 4; nt++) {
            const _Float16* q = Bw + (size_t)(ntile + nt * 16 + lr) * K + kk + hi * 16;
            v16h bv = *(const v16h*)q;
            #pragma unroll
            for (int mt = 0; mt < 2; mt++)
                acc[mt][nt] = __builtin_amdgcn_wmma_f32_16x16x32_f16(
                    false, a[mt].v, false, bv, (short)0, acc[mt][nt], false, false);
        }
    }

    #pragma unroll
    for (int mt = 0; mt < 2; mt++)
        #pragma unroll
        for (int nt = 0; nt < 4; nt++)
            #pragma unroll
            for (int r = 0; r < 8; r++) {
                int row = mtile + mt * 16 + r + hi * 8;
                int col = ntile + nt * 16 + lr;
                int b = row / (OH * OW);
                int rem = row % (OH * OW);
                int y = rem / OW, x = rem % OW;
                float vv = fmaxf(acc[mt][nt][r] + bias[col], 0.0f);
                if (MODE == 0)
                    out[((size_t)(b * PADHW + (y + 1) * PADW + (x + 1))) * CCH + col] = (_Float16)vv;
                else
                    out[((size_t)(b * 36 + y * 6 + x)) * CCH + col] = (_Float16)vv;
            }
}

// ---------------------------------------------------------------------------
// One-hot encode + im2col for conv1: A1[B*64][96], k = tap*8 + ic (72..95 zero)
// ---------------------------------------------------------------------------
__global__ void onehot_im2col(const int* __restrict__ boards, _Float16* __restrict__ A1)
{
    int m = blockIdx.x * blockDim.x + threadIdx.x;
    if (m >= BATCH * 64) return;
    int b = m >> 6, rem = m & 63, y = rem >> 3, x = rem & 7;
    _Float16* row = A1 + (size_t)m * 96;
    #pragma unroll
    for (int tap = 0; tap < 9; tap++) {
        int dy = tap / 3, dx = tap % 3;
        int iy = y + dy - 1, ix = x + dx - 1;
        int bv = 1000;
        if (iy >= 0 && iy < 8 && ix >= 0 && ix < 8) bv = boards[b * 64 + iy * 8 + ix];
        #pragma unroll
        for (int ic = 0; ic < 8; ic++) {
            int tgt = (ic < 4) ? (ic + 1) : -(ic - 3);
            row[tap * 8 + ic] = (_Float16)((bv == tgt) ? 1.0f : 0.0f);
        }
    }
    #pragma unroll
    for (int k = 72; k < 96; k++) row[k] = (_Float16)0.0f;
}

// ---------------------------------------------------------------------------
// Weight prep: conv OIHW f32 -> [OC][Kp] f16 with BN scale folded; bias folded
// ---------------------------------------------------------------------------
__global__ void prep_convw(const float* __restrict__ w, const float* __restrict__ cb,
                           const float* __restrict__ g, const float* __restrict__ bb,
                           const float* __restrict__ mm, const float* __restrict__ vv,
                           _Float16* __restrict__ wq, float* __restrict__ bias2,
                           int IC, int Kp)
{
    int idx = blockIdx.x * blockDim.x + threadIdx.x;
    if (idx >= 256 * Kp) return;
    int oc = idx / Kp, k = idx % Kp;
    float scale = g[oc] * rsqrtf(vv[oc] + BN_EPS);
    float val = 0.f;
    if (k < 9 * IC) {
        int tap = k / IC, ic = k % IC;
        int dy = tap / 3, dx = tap % 3;
        val = w[((size_t)(oc * IC + ic) * 3 + dy) * 3 + dx] * scale;
    }
    wq[(size_t)oc * Kp + k] = (_Float16)val;
    if (k == 0) bias2[oc] = (cb[oc] - mm[oc]) * scale + bb[oc];
}

// fc1 weights: permute K from (c*36 + s) to (s*256 + c), fold BN
__global__ void prep_fc1w(const float* __restrict__ w, const float* __restrict__ cb,
                          const float* __restrict__ g, const float* __restrict__ bb,
                          const float* __restrict__ mm, const float* __restrict__ vv,
                          _Float16* __restrict__ wq, float* __restrict__ bias2)
{
    const int N = 1024, K = 9216;
    int idx = blockIdx.x * blockDim.x + threadIdx.x;
    if (idx >= N * K) return;
    int n = idx / K, k = idx % K;       // our layout: k = s*256 + c
    int c = k & 255, s = k >> 8;        // s in [0,36)
    float scale = g[n] * rsqrtf(vv[n] + BN_EPS);
    wq[(size_t)n * K + k] = (_Float16)(w[(size_t)n * K + c * 36 + s] * scale);
    if (k == 0) bias2[n] = (cb[n] - mm[n]) * scale + bb[n];
}

// fc2 (BN folded) / fc3 (plain) weight prep
__global__ void prep_fcw(const float* __restrict__ w, const float* __restrict__ cb,
                         const float* __restrict__ g, const float* __restrict__ bb,
                         const float* __restrict__ mm, const float* __restrict__ vv,
                         _Float16* __restrict__ wq, float* __restrict__ bias2,
                         int N, int K, int has_bn)
{
    int idx = blockIdx.x * blockDim.x + threadIdx.x;
    if (idx >= N * K) return;
    int n = idx / K, k = idx % K;
    float scale = has_bn ? g[n] * rsqrtf(vv[n] + BN_EPS) : 1.0f;
    wq[(size_t)n * K + k] = (_Float16)(w[(size_t)n * K + k] * scale);
    if (k == 0) bias2[n] = has_bn ? (cb[n] - mm[n]) * scale + bb[n] : cb[n];
}

__global__ void fill_zero8(_Float16* __restrict__ p, long n8)
{
    long i = (long)blockIdx.x * blockDim.x + threadIdx.x;
    if (i < n8) {
        v8h z = {(_Float16)0,(_Float16)0,(_Float16)0,(_Float16)0,
                 (_Float16)0,(_Float16)0,(_Float16)0,(_Float16)0};
        ((v8h*)p)[i] = z;
    }
}

// value head: v[b] = tanh(dot(fc2out[b,:512], fc4_w) + fc4_b)
__global__ __launch_bounds__(32)
void fc4_kernel(const _Float16* __restrict__ x, const float* __restrict__ w,
                const float* __restrict__ b, float* __restrict__ outv)
{
    int row = blockIdx.x, lane = threadIdx.x;
    float s = 0.f;
    for (int k = lane; k < 512; k += 32)
        s += (float)x[(size_t)row * 512 + k] * w[k];
    #pragma unroll
    for (int o = 16; o > 0; o >>= 1) s += __shfl_xor(s, o, 32);
    if (lane == 0) outv[row] = tanhf(s + b[0]);
}

// in-place log-softmax over rows of 512 f32
__global__ __launch_bounds__(256)
void log_softmax_rows(float* __restrict__ pi)
{
    __shared__ float sm[256];
    int row = blockIdx.x, t = threadIdx.x;
    float* p = pi + (size_t)row * 512;
    float a0 = p[t], a1 = p[t + 256];
    sm[t] = fmaxf(a0, a1);
    __syncthreads();
    for (int o = 128; o > 0; o >>= 1) { if (t < o) sm[t] = fmaxf(sm[t], sm[t + o]); __syncthreads(); }
    float mx = sm[0];
    __syncthreads();
    sm[t] = expf(a0 - mx) + expf(a1 - mx);
    __syncthreads();
    for (int o = 128; o > 0; o >>= 1) { if (t < o) sm[t] += sm[t + o]; __syncthreads(); }
    float lse = logf(sm[0]) + mx;
    p[t] = a0 - lse;
    p[t + 256] = a1 - lse;
}

// ---------------------------------------------------------------------------
static char* carve(char*& p, size_t bytes)
{
    char* r = p;
    p += (bytes + 255) & ~(size_t)255;
    return r;
}

extern "C" void kernel_launch(void* const* d_in, const int* in_sizes, int n_in,
                              void* d_out, int out_size, void* d_ws, size_t ws_size,
                              hipStream_t stream)
{
    (void)in_sizes; (void)n_in; (void)out_size; (void)ws_size;

    const int* boards = (const int*)d_in[0];
    const float* conv_w[4] = {(const float*)d_in[1], (const float*)d_in[3],
                              (const float*)d_in[5], (const float*)d_in[7]};
    const float* conv_b[4] = {(const float*)d_in[2], (const float*)d_in[4],
                              (const float*)d_in[6], (const float*)d_in[8]};
    const float *bn_g[4], *bn_b[4], *bn_m[4], *bn_v[4];
    for (int i = 0; i < 4; i++) {
        bn_g[i] = (const float*)d_in[9  + 4 * i];
        bn_b[i] = (const float*)d_in[10 + 4 * i];
        bn_m[i] = (const float*)d_in[11 + 4 * i];
        bn_v[i] = (const float*)d_in[12 + 4 * i];
    }
    const float* fc1_w   = (const float*)d_in[25]; const float* fc1_b   = (const float*)d_in[26];
    const float* fcbn1_g = (const float*)d_in[27]; const float* fcbn1_b = (const float*)d_in[28];
    const float* fcbn1_m = (const float*)d_in[29]; const float* fcbn1_v = (const float*)d_in[30];
    const float* fc2_w   = (const float*)d_in[31]; const float* fc2_b   = (const float*)d_in[32];
    const float* fcbn2_g = (const float*)d_in[33]; const float* fcbn2_b = (const float*)d_in[34];
    const float* fcbn2_m = (const float*)d_in[35]; const float* fcbn2_v = (const float*)d_in[36];
    const float* fc3_w   = (const float*)d_in[37]; const float* fc3_b   = (const float*)d_in[38];
    const float* fc4_w   = (const float*)d_in[39]; const float* fc4_b   = (const float*)d_in[40];

    // workspace carve
    char* p = (char*)d_ws;
    _Float16* act0 = (_Float16*)carve(p, (size_t)BATCH * PADHW * CCH * 2);
    _Float16* act1 = (_Float16*)carve(p, (size_t)BATCH * PADHW * CCH * 2);
    _Float16* A1   = (_Float16*)carve(p, (size_t)BATCH * 64 * 96 * 2);
    _Float16* c4o  = (_Float16*)carve(p, (size_t)BATCH * 36 * CCH * 2);
    _Float16* f1o  = (_Float16*)carve(p, (size_t)BATCH * 1024 * 2);
    _Float16* f2o  = (_Float16*)carve(p, (size_t)BATCH * 512 * 2);
    _Float16* wq1  = (_Float16*)carve(p, (size_t)256 * 96 * 2);
    _Float16* wq2  = (_Float16*)carve(p, (size_t)256 * 2304 * 2);
    _Float16* wq3  = (_Float16*)carve(p, (size_t)256 * 2304 * 2);
    _Float16* wq4  = (_Float16*)carve(p, (size_t)256 * 2304 * 2);
    _Float16* fq1  = (_Float16*)carve(p, (size_t)1024 * 9216 * 2);
    _Float16* fq2  = (_Float16*)carve(p, (size_t)512 * 1024 * 2);
    _Float16* fq3  = (_Float16*)carve(p, (size_t)512 * 512 * 2);
    float* bias_c1 = (float*)carve(p, 256 * 4);
    float* bias_c2 = (float*)carve(p, 256 * 4);
    float* bias_c3 = (float*)carve(p, 256 * 4);
    float* bias_c4 = (float*)carve(p, 256 * 4);
    float* bias_f1 = (float*)carve(p, 1024 * 4);
    float* bias_f2 = (float*)carve(p, 512 * 4);
    float* bias_f3 = (float*)carve(p, 512 * 4);

    float* pi_out = (float*)d_out;                      // [2048,512] logits -> log_softmax
    float* v_out  = pi_out + (size_t)BATCH * 512;       // [2048]

    // 1) zero padded activation buffers (borders must be 0 every call)
    {
        long n8 = (long)BATCH * PADHW * CCH / 8;
        fill_zero8<<<dim3((n8 + 255) / 256), dim3(256), 0, stream>>>(act0, n8);
        fill_zero8<<<dim3((n8 + 255) / 256), dim3(256), 0, stream>>>(act1, n8);
    }

    // 2) weight prep (BN folded)
    prep_convw<<<dim3((256 * 96 + 255) / 256), dim3(256), 0, stream>>>(
        conv_w[0], conv_b[0], bn_g[0], bn_b[0], bn_m[0], bn_v[0], wq1, bias_c1, 8, 96);
    prep_convw<<<dim3((256 * 2304 + 255) / 256), dim3(256), 0, stream>>>(
        conv_w[1], conv_b[1], bn_g[1], bn_b[1], bn_m[1], bn_v[1], wq2, bias_c2, 256, 2304);
    prep_convw<<<dim3((256 * 2304 + 255) / 256), dim3(256), 0, stream>>>(
        conv_w[2], conv_b[2], bn_g[2], bn_b[2], bn_m[2], bn_v[2], wq3, bias_c3, 256, 2304);
    prep_convw<<<dim3((256 * 2304 + 255) / 256), dim3(256), 0, stream>>>(
        conv_w[3], conv_b[3], bn_g[3], bn_b[3], bn_m[3], bn_v[3], wq4, bias_c4, 256, 2304);
    prep_fc1w<<<dim3((1024 * 9216 + 255) / 256), dim3(256), 0, stream>>>(
        fc1_w, fc1_b, fcbn1_g, fcbn1_b, fcbn1_m, fcbn1_v, fq1, bias_f1);
    prep_fcw<<<dim3((512 * 1024 + 255) / 256), dim3(256), 0, stream>>>(
        fc2_w, fc2_b, fcbn2_g, fcbn2_b, fcbn2_m, fcbn2_v, fq2, bias_f2, 512, 1024, 1);
    prep_fcw<<<dim3((512 * 512 + 255) / 256), dim3(256), 0, stream>>>(
        fc3_w, fc3_b, nullptr, nullptr, nullptr, nullptr, fq3, bias_f3, 512, 512, 0);

    // 3) one-hot + im2col for conv1
    onehot_im2col<<<dim3(BATCH * 64 / 256), dim3(256), 0, stream>>>(boards, A1);

    // 4) conv1: explicit GEMM [131072 x 96] * [256 x 96]^T -> padded act0 interior
    gemm_f16<1><<<dim3(BATCH * 64 / 32, 256 / 64), dim3(32), 0, stream>>>(
        A1, wq1, bias_c1, act0, 256, 96);

    // 5) conv2/conv3: implicit GEMM, SAME, padded <-> padded ping-pong
    conv_gemm<0><<<dim3(BATCH * 64 / 32, 256 / 64), dim3(32), 0, stream>>>(act0, wq2, bias_c2, act1);
    conv_gemm<0><<<dim3(BATCH * 64 / 32, 256 / 64), dim3(32), 0, stream>>>(act1, wq3, bias_c3, act0);

    // 6) conv4: VALID -> compact [B][36][256] (== fc1 input, k = (y*6+x)*256+c)
    conv_gemm<1><<<dim3(BATCH * 36 / 32, 256 / 64), dim3(32), 0, stream>>>(act0, wq4, bias_c4, c4o);

    // 7) fc1 / fc2 (f16 relu), fc3 (f32 logits into d_out)
    gemm_f16<0><<<dim3(BATCH / 32, 1024 / 64), dim3(32), 0, stream>>>(c4o, fq1, bias_f1, f1o, 1024, 9216);
    gemm_f16<0><<<dim3(BATCH / 32,  512 / 64), dim3(32), 0, stream>>>(f1o, fq2, bias_f2, f2o,  512, 1024);
    gemm_f16<2><<<dim3(BATCH / 32,  512 / 64), dim3(32), 0, stream>>>(f2o, fq3, bias_f3, pi_out, 512, 512);

    // 8) value head + policy log-softmax
    fc4_kernel<<<dim3(BATCH), dim3(32), 0, stream>>>(f2o, fc4_w, fc4_b, v_out);
    log_softmax_rows<<<dim3(BATCH), dim3(256), 0, stream>>>(pi_out);
}
